// CombinedLoss_84361747628734
// MI455X (gfx1250) — compile-verified, matrix operand order checked by reading, not run on hardware
//
#include <hip/hip_runtime.h>
#include <hip/hip_bf16.h>

// ---------------------------------------------------------------------------
// CombinedLoss for MI455X (gfx1250, wave32).
//   inputs : f32 [8,2,256,256]   targets : int [8,256,256]   out: f32 scalar
//
// Pipeline:
//   kStats   : elementwise CE / probs / focal-CE; per-block partial sums
//   kReduceA : combine partials -> 5 global sums + per-batch fg counts
//   kEdt1/2  : exact squared EDT (separable min-plus) -> sum of boundary wts
//   kRowBox/kColBox : 7x7 box-sum of t via V_WMMA_F32_16X16X32_F16
//                     (C = S * (T * S), S = banded ones matrix, exact in f16;
//                      intermediate stored TRANSPOSED so both stages use
//                      contiguous 128-bit fragment loads/stores)
//   kFinal   : assemble scalar loss
// ---------------------------------------------------------------------------

typedef _Float16 v16h __attribute__((ext_vector_type(16)));
typedef _Float16 v8h  __attribute__((ext_vector_type(8)));
typedef float    v8f  __attribute__((ext_vector_type(8)));

#define B_  8
#define H_  256
#define W_  256
#define HW_ 65536          // H_*W_
#define N_  524288         // B_*H_*W_
#define NBLK 2048          // N_/256

__device__ __forceinline__ float blockReduceSum256(float v, float* sm) {
  __syncthreads();
  int lane = threadIdx.x & 31;
  int wid  = threadIdx.x >> 5;
#pragma unroll
  for (int o = 16; o > 0; o >>= 1) v += __shfl_xor(v, o, 32);
  if (lane == 0) sm[wid] = v;
  __syncthreads();
  float r = 0.0f;
  if (wid == 0) {
    r = (lane < 8) ? sm[lane] : 0.0f;
#pragma unroll
    for (int o = 4; o > 0; o >>= 1) r += __shfl_xor(r, o, 32);
  }
  return r;  // valid in thread 0
}

// ---------------- elementwise statistics ------------------------------------
__global__ __launch_bounds__(256) void kStats(const float* __restrict__ x,
                                              const int* __restrict__ tgt,
                                              float* __restrict__ probs,
                                              _Float16* __restrict__ tf16,
                                              float* __restrict__ partials) {
  __shared__ float sm[8];
  int idx = blockIdx.x * 256 + threadIdx.x;     // < N_
  int b   = idx >> 16;
  int hw  = idx & (HW_ - 1);
  float x0 = x[b * (2 * HW_) + hw];
  float x1 = x[b * (2 * HW_) + HW_ + hw];
  int   tg = tgt[idx];
  // 2-class log-softmax
  float m   = fmaxf(x0, x1);
  float lse = m + logf(expf(x0 - m) + expf(x1 - m));
  float ce  = lse - (tg == 1 ? x1 : x0);        // -log p_target
  float p   = expf(x1 - lse);                   // class-1 prob
  float t   = (tg == 1) ? 1.0f : 0.0f;
  probs[idx] = p;
  tf16[idx]  = (_Float16)t;
  // focal BCE term on z = sigmoid(10*(p-0.5)), tb = sigmoid(10*(t-0.5))
  float z   = 1.0f / (1.0f + expf(-10.0f * (p - 0.5f)));
  float tb  = (tg == 1) ? 0.99330714907571527f : 0.00669285092428486f;
  float bce = log1pf(expf(z)) - tb * z;         // logaddexp(0,z) - tb*z
  float omp = 1.0f - expf(-bce);
  float fce = 0.25f * omp * omp * bce;          // ALPHA*(1-ptb)^GAMMA*bce

  float s_ce = blockReduceSum256(ce,    sm);
  float s_p  = blockReduceSum256(p,     sm);
  float s_t  = blockReduceSum256(t,     sm);
  float s_pt = blockReduceSum256(p * t, sm);
  float s_f  = blockReduceSum256(fce,   sm);
  if (threadIdx.x == 0) {
    float* row = partials + blockIdx.x * 5;
    row[0] = s_ce; row[1] = s_p; row[2] = s_t; row[3] = s_pt; row[4] = s_f;
  }
}

// ---------------- combine partials ------------------------------------------
__global__ __launch_bounds__(256) void kReduceA(const float* __restrict__ partials,
                                                float* __restrict__ sc) {
  __shared__ float sm[8];
  float a0 = 0, a1 = 0, a2 = 0, a3 = 0, a4 = 0;
  for (int j = 0; j < 8; j++) {
    const float* row = partials + (threadIdx.x + 256 * j) * 5;
    a0 += row[0]; a1 += row[1]; a2 += row[2]; a3 += row[3]; a4 += row[4];
  }
  float s0 = blockReduceSum256(a0, sm);
  float s1 = blockReduceSum256(a1, sm);
  float s2 = blockReduceSum256(a2, sm);
  float s3 = blockReduceSum256(a3, sm);
  float s4 = blockReduceSum256(a4, sm);
  if (threadIdx.x == 0) { sc[0]=s0; sc[1]=s1; sc[2]=s2; sc[3]=s3; sc[4]=s4; }
  __syncthreads();
  // per-batch foreground counts (t-sum): batch b owns partial rows [256b,256b+256)
  if (threadIdx.x < B_) {
    float c = 0.0f;
    for (int r = 0; r < 256; r++) c += partials[(threadIdx.x * 256 + r) * 5 + 2];
    sc[5 + threadIdx.x] = c;
  }
}

// ---------------- exact squared EDT, pass 1 (along H) -----------------------
__global__ __launch_bounds__(256) void kEdt1(const int* __restrict__ tgt,
                                             float* __restrict__ g) {
  __shared__ float f[H_];
  int b = blockIdx.x >> 8, w = blockIdx.x & 255;
  int j = threadIdx.x;
  f[j] = (tgt[b * HW_ + j * W_ + w] == 1) ? 1.0e12f : 0.0f;
  __syncthreads();
  float h  = (float)threadIdx.x;
  float mn = 3.0e12f;
  for (int jj = 0; jj < H_; jj++) {
    float d = h - (float)jj;
    mn = fminf(mn, fmaf(d, d, f[jj]));
  }
  g[b * HW_ + threadIdx.x * W_ + w] = mn;
}

// ---------------- EDT pass 2 (along W) + boundary-weight row sums -----------
__global__ __launch_bounds__(256) void kEdt2(const float* __restrict__ g,
                                             float* __restrict__ bw_parts) {
  __shared__ float ga[W_];
  __shared__ float sm[8];
  int b = blockIdx.x >> 8, h = blockIdx.x & 255;
  ga[threadIdx.x] = g[b * HW_ + h * W_ + threadIdx.x];
  __syncthreads();
  float w  = (float)threadIdx.x;
  float mn = 3.0e12f;
  for (int ww = 0; ww < W_; ww++) {
    float d = w - (float)ww;
    mn = fminf(mn, fmaf(d, d, ga[ww]));
  }
  float bwv = 10.0f * expf(-mn * (1.0f / 98.0f));   // W*exp(-d2/(2*sigma^2))
  float s = blockReduceSum256(bwv, sm);
  if (threadIdx.x == 0) bw_parts[blockIdx.x] = s;
}

// ---------------- box-sum stage 1: C1 = T * S, stored transposed ------------
__global__ __launch_bounds__(32) void kRowBox(const _Float16* __restrict__ tf16,
                                              _Float16* __restrict__ c1t) {
  int tile = blockIdx.x;                 // b*256 + mtile*16 + ntile
  int b  = tile >> 8;
  int m0 = ((tile >> 4) & 15) * 16;      // output row block (h)
  int n0 = (tile & 15) * 16;             // output col block (w)
  int lane = threadIdx.x;
  int mr = lane & 15;
  int kb = (lane >> 4) * 8;              // ISA A-layout K sub-block
  const _Float16* Abase = tf16 + b * HW_ + (m0 + mr) * W_;
  int n = n0 + mr;                       // B-fragment column for this lane
  v8f acc = {};
  for (int k0 = 0; k0 < W_; k0 += 32) {
    v8h lo = *(const v8h*)(Abase + k0 + kb);          // 128-bit loads
    v8h hi = *(const v8h*)(Abase + k0 + 16 + kb);
    v16h a, bb;
#pragma unroll
    for (int i = 0; i < 8; i++) {
      a[i]     = lo[i];
      a[8 + i] = hi[i];
      int k  = k0 + kb + i;
      int k2 = k0 + 16 + kb + i;
      int d1 = k  - n; if (d1 < 0) d1 = -d1;
      int d2 = k2 - n; if (d2 < 0) d2 = -d2;
      bb[i]     = (_Float16)(d1 <= 3 ? 1.0f : 0.0f);   // S[k][n] in registers
      bb[8 + i] = (_Float16)(d2 <= 3 ? 1.0f : 0.0f);
    }
    acc = __builtin_amdgcn_wmma_f32_16x16x32_f16(false, a, false, bb,
                                                 (short)0, acc, false, false);
  }
  // C/D layout: VGPR r -> M = r (+8 for lanes 16..31), N = lane&15.
  // Lane holds 8 consecutive rows of one column -> contiguous in the
  // transposed layout c1t[w][h]: a single 128-bit store per lane.
  int nn = n0 + (lane & 15);
  int mb = m0 + (lane >> 4) * 8;
  v8h o;
#pragma unroll
  for (int r = 0; r < 8; r++) o[r] = (_Float16)acc[r];  // values 0..7: exact
  *(v8h*)(c1t + b * HW_ + nn * H_ + mb) = o;
}

// ---------------- box-sum stage 2: C2 = S * C1; count boundary pixels -------
__global__ __launch_bounds__(32) void kColBox(const _Float16* __restrict__ c1t,
                                              float* __restrict__ gb_parts) {
  int tile = blockIdx.x;
  int b  = tile >> 8;
  int m0 = ((tile >> 4) & 15) * 16;
  int n0 = (tile & 15) * 16;
  int lane = threadIdx.x;
  int mr = lane & 15;
  int kb = (lane >> 4) * 8;
  int nn = n0 + mr;
  // B[k][n] = C1[k][nn] = c1t[nn][k]: contiguous in k -> 128-bit loads
  const _Float16* Bbase = c1t + b * HW_ + nn * H_;
  v8f acc = {};
  for (int k0 = 0; k0 < H_; k0 += 32) {
    v8h lo = *(const v8h*)(Bbase + k0 + kb);
    v8h hi = *(const v8h*)(Bbase + k0 + 16 + kb);
    v16h a, bb;
#pragma unroll
    for (int i = 0; i < 8; i++) {
      bb[i]     = lo[i];
      bb[8 + i] = hi[i];
      int k  = k0 + kb + i;
      int k2 = k0 + 16 + kb + i;
      int d1 = k  - (m0 + mr); if (d1 < 0) d1 = -d1;
      int d2 = k2 - (m0 + mr); if (d2 < 0) d2 = -d2;
      a[i]     = (_Float16)(d1 <= 3 ? 1.0f : 0.0f);   // S[m][k] in registers
      a[8 + i] = (_Float16)(d2 <= 3 ? 1.0f : 0.0f);
    }
    acc = __builtin_amdgcn_wmma_f32_16x16x32_f16(false, a, false, bb,
                                                 (short)0, acc, false, false);
  }
  // gb = (0 < boxsum < 49); boxsum is an exact small integer here
  float s = 0.0f;
#pragma unroll
  for (int r = 0; r < 8; r++) {
    float v = acc[r];
    s += (v > 0.5f && v < 48.5f) ? 1.0f : 0.0f;
  }
#pragma unroll
  for (int o = 16; o > 0; o >>= 1) s += __shfl_xor(s, o, 32);
  if (lane == 0) gb_parts[tile] = s;
}

// ---------------- final combine ---------------------------------------------
__global__ __launch_bounds__(256) void kFinal(const float* __restrict__ sc,
                                              const float* __restrict__ bw_parts,
                                              const float* __restrict__ gb_parts,
                                              float* __restrict__ out) {
  __shared__ float sm[8];
  float bwl = 0.0f, gbl = 0.0f;
  for (int j = 0; j < 8; j++) {
    int p = threadIdx.x + 256 * j;           // < NBLK
    int b = p >> 8;
    float gate = (sc[5 + b] > 0.0f) ? 1.0f : 0.0f;   // has_fg per batch
    bwl += bw_parts[p] * gate;
    gbl += gb_parts[p];
  }
  float bw_sum = blockReduceSum256(bwl, sm);
  float gb_sum = blockReduceSum256(gbl, sm);
  if (threadIdx.x != 0) return;

  const float N  = (float)N_;
  const float SM = 1e-6f;
  float ce_sum = sc[0], p_sum = sc[1], t_sum = sc[2], inter = sc[3], fce_sum = sc[4];

  float ce   = ce_sum / N;
  float dice = 1.0f - (2.0f * inter + SM) / (p_sum + t_sum + SM);
  float pt   = expf(-ce);
  float focal = 0.25f * (1.0f - pt) * (1.0f - pt) * ce;        // ALPHA,(1-pt)^GAMMA
  float fp = p_sum - inter, fn = t_sum - inter;
  float tversky = 1.0f - inter / (inter + 0.7f * fp + 0.3f * fn);
  float ftv = tversky * tversky;                               // ^(1/DELTA)
  // wfce: class weights normalize to sum 1 -> only sum(bw) matters
  float focal_ce = fce_sum / N;
  float wfce = focal_ce * bw_sum / (8.0f * N);                 // B*B*H*W = 8N
  // pb == 1 everywhere (sigmoid(probs) in (0.5,0.7311) -> boxsum in (0,49))
  float bndice = 1.0f - (2.0f * gb_sum + SM) / (N + gb_sum + SM);

  out[0] = 2.0f * dice + focal + ce + ftv + wfce + bndice;
}

// ---------------------------------------------------------------------------
extern "C" void kernel_launch(void* const* d_in, const int* in_sizes, int n_in,
                              void* d_out, int out_size, void* d_ws, size_t ws_size,
                              hipStream_t stream) {
  (void)in_sizes; (void)n_in; (void)out_size; (void)ws_size;
  const float* x   = (const float*)d_in[0];
  const int*   tgt = (const int*)d_in[1];
  float*       out = (float*)d_out;

  char* ws = (char*)d_ws;
  float*    probs    = (float*)(ws + 0);                      // 2 MiB
  float*    g        = (float*)(ws + 2097152);                // 2 MiB
  _Float16* tf16     = (_Float16*)(ws + 4194304);             // 1 MiB
  _Float16* c1t      = (_Float16*)(ws + 5242880);             // 1 MiB (transposed)
  float*    partials = (float*)(ws + 6291456);                // 40 KiB
  float*    bw_parts = (float*)(ws + 6332416);                // 8 KiB
  float*    gb_parts = (float*)(ws + 6340608);                // 8 KiB
  float*    sc       = (float*)(ws + 6348800);                // 64 B

  kStats  <<<NBLK, 256, 0, stream>>>(x, tgt, probs, tf16, partials);
  kReduceA<<<1,    256, 0, stream>>>(partials, sc);
  kEdt1   <<<NBLK, 256, 0, stream>>>(tgt, g);
  kEdt2   <<<NBLK, 256, 0, stream>>>(g, bw_parts);
  kRowBox <<<NBLK,  32, 0, stream>>>(tf16, c1t);
  kColBox <<<NBLK,  32, 0, stream>>>(c1t, gb_parts);
  kFinal  <<<1,    256, 0, stream>>>(sc, bw_parts, gb_parts, out);
}
